// ScaledDotProductAttention_68822555951318
// MI455X (gfx1250) — matrix operand order repaired and checked by reading, not compiled, not run on hardware
//
#include <hip/hip_runtime.h>

typedef _Float16 half_t;
typedef __attribute__((ext_vector_type(16))) _Float16 v16h;
typedef __attribute__((ext_vector_type(8)))  _Float16 v8h;
typedef __attribute__((ext_vector_type(4)))  _Float16 v4h;
typedef __attribute__((ext_vector_type(8)))  float    v8f;

constexpr int   Bz = 2, Sq = 2048, Hh = 16, Dd = 64;
constexpr int   Bc  = 32;                // keys per tile (legacy kernel)
constexpr int   Bc2 = 64;                // keys per tile (fast kernel)
// 1/sqrt(64) * log2(e): scores come out of the QK WMMA in log2 domain,
// so softmax uses the native v_exp_f32 (exp2) with no extra multiplies.
constexpr float SCALE_L2E = 0.125f * 1.44269504088896f;
constexpr float NEGINF = -1e20f;
constexpr size_t KV_ELEMS = (size_t)Bz * Hh * Sq * Dd;   // 4,194,304 per tensor

__device__ __forceinline__ float fast_exp2(float x) {
  return __builtin_amdgcn_exp2f(x);
}

// 16-lane butterfly max via DPP (row-confined; xor-basis {1,2,7,15})
template <int CTRL>
__device__ __forceinline__ float dpp_fmax(float x) {
  const int y = __builtin_amdgcn_update_dpp(0, __float_as_int(x), CTRL, 0xF, 0xF, true);
  return fmaxf(x, __int_as_float(y));
}
#define DPP_QUAD_XOR1 0xB1   // quad_perm [1,0,3,2]
#define DPP_QUAD_XOR2 0x4E   // quad_perm [2,3,0,1]
#define DPP_HALF_MIRR 0x141  // row_half_mirror (xor 7)
#define DPP_ROW_MIRR  0x140  // row_mirror      (xor 15)

__device__ __forceinline__ float rowmax16(float x) {
  x = dpp_fmax<DPP_QUAD_XOR1>(x);
  x = dpp_fmax<DPP_QUAD_XOR2>(x);
  x = dpp_fmax<DPP_HALF_MIRR>(x);
  x = dpp_fmax<DPP_ROW_MIRR>(x);
  return x;
}

// ---------------------------------------------------------------------------
// Prep 1: Kh[b][h][s][d] = (f16) K[b][s][h][d]   (coalesced read & write)
// ---------------------------------------------------------------------------
__global__ __launch_bounds__(256) void prep_k_kernel(
    const float* __restrict__ K, half_t* __restrict__ Kh)
{
  const size_t i = (size_t)blockIdx.x * blockDim.x + threadIdx.x;
  const int d = i & 63;
  const int s = (i >> 6) & 2047;
  const int h = (i >> 17) & 15;
  const int b = i >> 21;
  Kh[i] = (half_t)K[(((size_t)b * Sq + s) * Hh + h) * Dd + d];
}

// ---------------------------------------------------------------------------
// Prep 2: Vh[b][h][d][slot] = (f16) V[b][s][h][d], transposed via LDS tile.
// Within each 64-key group, slot sl holds physical key ((sl&3)*16 + (sl>>2)),
// matching the fast kernel's packed P-tile store (slot = 4*ln + g).
// ---------------------------------------------------------------------------
__global__ __launch_bounds__(256) void prep_v_kernel(
    const float* __restrict__ V, half_t* __restrict__ Vh)
{
  __shared__ half_t t[64 * 65];
  const int tid = threadIdx.x;
  const int b = blockIdx.z, h = blockIdx.y;
  const int s0 = blockIdx.x * 64;          // one full 64-key group per block
  #pragma unroll
  for (int it = 0; it < 16; ++it) {        // read: d fastest -> coalesced
    const int idx = it * 256 + tid;
    const int d = idx & 63, s = idx >> 6;
    t[d * 65 + s] = (half_t)V[(((size_t)b * Sq + (s0 + s)) * Hh + h) * Dd + d];
  }
  __syncthreads();
  #pragma unroll
  for (int it = 0; it < 16; ++it) {        // write: slot fastest -> coalesced
    const int idx = it * 256 + tid;
    const int sl = idx & 63, d = idx >> 6;
    const int src = ((sl & 3) << 4) + (sl >> 2);   // de-interleave inside group
    Vh[(((size_t)b * Hh + h) * Dd + d) * Sq + (s0 + sl)] = t[d * 65 + src];
  }
}

// ---------------------------------------------------------------------------
// Flash attention, f16 K/V in fragment-native layouts, Bc=64 (fast path)
// ---------------------------------------------------------------------------
__global__ __launch_bounds__(128) void fa_fwd_f16kv_kernel(
    const float* __restrict__ Q, const half_t* __restrict__ Kh,
    const half_t* __restrict__ Vh, const int* __restrict__ Mk,
    float* __restrict__ O)
{
  __shared__ alignas(32) half_t Ps[4][16 * Bc2];  // per-wave P tile [row][slot]

  const int w    = threadIdx.x >> 5;
  const int lane = threadIdx.x & 31;
  const int hi   = lane >> 4;
  const int ln   = lane & 15;

  const int b  = blockIdx.z;
  const int h  = blockIdx.y;
  const int q0 = blockIdx.x * 64 + w * 16;

  half_t* psm = Ps[w];

  // Q as two 16x32 f16 A-fragments, pre-scaled by log2(e)/sqrt(d)
  const float* qrow = Q + (((size_t)b * Sq + (q0 + ln)) * Hh + h) * Dd;
  v16h aQ0, aQ1;
  #pragma unroll
  for (int e = 0; e < 16; ++e) {
    const int d = e + hi * 16;   // contiguous k order (matches B)
    aQ0[e] = (half_t)(qrow[d]      * SCALE_L2E);
    aQ1[e] = (half_t)(qrow[d + 32] * SCALE_L2E);
  }

  v16h bOne;                         // ones B-fragment for row-sum WMMA
  #pragma unroll
  for (int e = 0; e < 16; ++e) bOne[e] = (half_t)1.0f;

  const half_t* khb = Kh + ((size_t)b * Hh + h) * Sq * Dd;  // [s][d]
  const half_t* vhb = Vh + ((size_t)b * Hh + h) * Dd * Sq;  // [d][slot]

  v8f acc0 = {}, acc1 = {}, acc2 = {}, acc3 = {}, accL = {};
  float mrow[8];
  #pragma unroll
  for (int r = 0; r < 8; ++r) mrow[r] = -1e30f;

  const int* mb = Mk + (size_t)b * Sq;
  const int  nTiles = Sq / Bc2;

  for (int j = 0; j < nTiles; ++j) {
    const int k0 = j * Bc2;

    // ---- scores for 4 key groups of 16 (K fragments direct from global) ----
    v8f sg[4];
    int mk[4];
    #pragma unroll
    for (int g = 0; g < 4; ++g) {
      const half_t* kr = khb + (size_t)(k0 + g * 16 + ln) * Dd;
      const v16h bk0 = *(const v16h*)(kr + hi * 16);
      const v16h bk1 = *(const v16h*)(kr + 32 + hi * 16);
      v8f z = {};
      sg[g] = __builtin_amdgcn_wmma_f32_16x16x32_f16(false, aQ0, false, bk0, (short)0, z,     false, false);
      sg[g] = __builtin_amdgcn_wmma_f32_16x16x32_f16(false, aQ1, false, bk1, (short)0, sg[g], false, false);
      mk[g] = mb[k0 + g * 16 + ln];
    }

    if (j + 1 < nTiles) {
      __builtin_prefetch(khb + (size_t)(k0 + Bc2 + ln) * Dd, 0, 0);
      __builtin_prefetch(vhb + (size_t)ln * Sq + k0 + Bc2, 0, 0);
    }

    // ---- online softmax (log2 domain): lane holds rows r+8*hi ----
    #pragma unroll
    for (int r = 0; r < 8; ++r) {
      float a[4];
      #pragma unroll
      for (int g = 0; g < 4; ++g) a[g] = mk[g] ? sg[g][r] : NEGINF;
      const float rm = rowmax16(fmaxf(fmaxf(a[0], a[1]), fmaxf(a[2], a[3])));
      const float mn = fmaxf(mrow[r], rm);
      const float alpha = fast_exp2(mrow[r] - mn);
      mrow[r] = mn;
      v4h pk;
      #pragma unroll
      for (int g = 0; g < 4; ++g)
        pk[g] = (half_t)(mk[g] ? fast_exp2(a[g] - mn) : 0.f);
      acc0[r] *= alpha; acc1[r] *= alpha; acc2[r] *= alpha; acc3[r] *= alpha;
      accL[r] *= alpha;
      // packed b64 store: slots 4*ln .. 4*ln+3 of row r+8*hi
      *(v4h*)(psm + (r + 8 * hi) * Bc2 + 4 * ln) = pk;
    }

    asm volatile("s_wait_dscnt 0" ::: "memory");

    // ---- reload P as two A fragments (slots 0..31 and 32..63) ----
    const half_t* pp = psm + ln * Bc2 + hi * 16;
    const v8h pa0 = *(const v8h*)pp;
    const v8h pa1 = *(const v8h*)(pp + 8);
    const v8h pb0 = *(const v8h*)(pp + 32);
    const v8h pb1 = *(const v8h*)(pp + 40);
    const v16h aPa = __builtin_shufflevector(pa0, pa1,
        0, 1, 2, 3, 4, 5, 6, 7, 8, 9, 10, 11, 12, 13, 14, 15);
    const v16h aPb = __builtin_shufflevector(pb0, pb1,
        0, 1, 2, 3, 4, 5, 6, 7, 8, 9, 10, 11, 12, 13, 14, 15);

    // ---- V B-fragments (interleaved slots) and PV accumulation ----
    const half_t* vr0 = vhb + (size_t)(0 * 16 + ln) * Sq + k0 + hi * 16;
    const half_t* vr1 = vhb + (size_t)(1 * 16 + ln) * Sq + k0 + hi * 16;
    const half_t* vr2 = vhb + (size_t)(2 * 16 + ln) * Sq + k0 + hi * 16;
    const half_t* vr3 = vhb + (size_t)(3 * 16 + ln) * Sq + k0 + hi * 16;
    const v16h bva0 = *(const v16h*)vr0;
    const v16h bvb0 = *(const v16h*)(vr0 + 32);
    const v16h bva1 = *(const v16h*)vr1;
    const v16h bvb1 = *(const v16h*)(vr1 + 32);
    const v16h bva2 = *(const v16h*)vr2;
    const v16h bvb2 = *(const v16h*)(vr2 + 32);
    const v16h bva3 = *(const v16h*)vr3;
    const v16h bvb3 = *(const v16h*)(vr3 + 32);

    acc0 = __builtin_amdgcn_wmma_f32_16x16x32_f16(false, aPa, false, bva0, (short)0, acc0, false, false);
    acc0 = __builtin_amdgcn_wmma_f32_16x16x32_f16(false, aPb, false, bvb0, (short)0, acc0, false, false);
    acc1 = __builtin_amdgcn_wmma_f32_16x16x32_f16(false, aPa, false, bva1, (short)0, acc1, false, false);
    acc1 = __builtin_amdgcn_wmma_f32_16x16x32_f16(false, aPb, false, bvb1, (short)0, acc1, false, false);
    acc2 = __builtin_amdgcn_wmma_f32_16x16x32_f16(false, aPa, false, bva2, (short)0, acc2, false, false);
    acc2 = __builtin_amdgcn_wmma_f32_16x16x32_f16(false, aPb, false, bvb2, (short)0, acc2, false, false);
    acc3 = __builtin_amdgcn_wmma_f32_16x16x32_f16(false, aPa, false, bva3, (short)0, acc3, false, false);
    acc3 = __builtin_amdgcn_wmma_f32_16x16x32_f16(false, aPb, false, bvb3, (short)0, acc3, false, false);
    accL = __builtin_amdgcn_wmma_f32_16x16x32_f16(false, aPa, false, bOne, (short)0, accL, false, false);
    accL = __builtin_amdgcn_wmma_f32_16x16x32_f16(false, aPb, false, bOne, (short)0, accL, false, false);

    asm volatile("" ::: "memory");
  }

  #pragma unroll
  for (int r = 0; r < 8; ++r) {
    const int   row = r + 8 * hi;
    const float l   = accL[r];
    const float inv = l > 0.f ? 1.f / l : 0.f;
    float* op = O + (((size_t)b * Sq + (q0 + row)) * Hh + h) * Dd;
    op[0 * 16 + ln] = acc0[r] * inv;
    op[1 * 16 + ln] = acc1[r] * inv;
    op[2 * 16 + ln] = acc2[r] * inv;
    op[3 * 16 + ln] = acc3[r] * inv;
  }
}

// ---------------------------------------------------------------------------
// Legacy fully-fused kernel (fallback when d_ws is too small)
// ---------------------------------------------------------------------------
__global__ __launch_bounds__(128) void fa_fwd_legacy_kernel(
    const float* __restrict__ Q, const float* __restrict__ K,
    const float* __restrict__ V, const int* __restrict__ Mk,
    float* __restrict__ O)
{
  __shared__ half_t Vs[4][Dd * Bc];
  __shared__ half_t Ps[4][16 * Bc];

  const int w = threadIdx.x >> 5, lane = threadIdx.x & 31;
  const int hi = lane >> 4, ln = lane & 15;
  const int b = blockIdx.z, h = blockIdx.y;
  const int q0 = blockIdx.x * 64 + w * 16;
  half_t* vsm = Vs[w];
  half_t* psm = Ps[w];

  const float* qrow = Q + (((size_t)b * Sq + (q0 + ln)) * Hh + h) * Dd;
  v16h aQ0, aQ1;
  #pragma unroll
  for (int e = 0; e < 16; ++e) {
    const int d = e + hi * 16;
    aQ0[e] = (half_t)(qrow[d]      * SCALE_L2E);
    aQ1[e] = (half_t)(qrow[d + 32] * SCALE_L2E);
  }
  v8f acc0 = {}, acc1 = {}, acc2 = {}, acc3 = {};
  float mrow[8], lrow[8];
  #pragma unroll
  for (int r = 0; r < 8; ++r) { mrow[r] = -1e30f; lrow[r] = 0.f; }
  const int* mb = Mk + (size_t)b * Sq;

  for (int j = 0; j < Sq / Bc; ++j) {
    const int k0 = j * Bc;
    const float* kp0 = K + (((size_t)b * Sq + (k0 + ln)) * Hh + h) * Dd;
    const float* kp1 = kp0 + (size_t)16 * Hh * Dd;
    v16h bK00, bK01, bK10, bK11;
    #pragma unroll
    for (int e = 0; e < 16; ++e) {
      bK00[e] = (half_t)kp0[hi * 16 + e];
      bK01[e] = (half_t)kp0[32 + hi * 16 + e];
      bK10[e] = (half_t)kp1[hi * 16 + e];
      bK11[e] = (half_t)kp1[32 + hi * 16 + e];
    }
    #pragma unroll
    for (int it = 0; it < 16; ++it) {
      const int fj = it * 32 + lane, key = fj >> 4, c4 = fj & 15;
      const float4 vv = *(const float4*)(V + (((size_t)b * Sq + (k0 + key)) * Hh + h) * Dd + c4 * 4);
      vsm[(c4 * 4 + 0) * Bc + key] = (half_t)vv.x;
      vsm[(c4 * 4 + 1) * Bc + key] = (half_t)vv.y;
      vsm[(c4 * 4 + 2) * Bc + key] = (half_t)vv.z;
      vsm[(c4 * 4 + 3) * Bc + key] = (half_t)vv.w;
    }
    v8f z = {};
    v8f s0 = __builtin_amdgcn_wmma_f32_16x16x32_f16(false, aQ0, false, bK00, (short)0, z,  false, false);
    s0     = __builtin_amdgcn_wmma_f32_16x16x32_f16(false, aQ1, false, bK01, (short)0, s0, false, false);
    v8f s1 = __builtin_amdgcn_wmma_f32_16x16x32_f16(false, aQ0, false, bK10, (short)0, z,  false, false);
    s1     = __builtin_amdgcn_wmma_f32_16x16x32_f16(false, aQ1, false, bK11, (short)0, s1, false, false);
    const int mk0 = mb[k0 + ln], mk1 = mb[k0 + 16 + ln];
    #pragma unroll
    for (int r = 0; r < 8; ++r) {
      const float a0 = mk0 ? s0[r] : NEGINF;
      const float a1 = mk1 ? s1[r] : NEGINF;
      const float rm = rowmax16(fmaxf(a0, a1));
      const float mn = fmaxf(mrow[r], rm);
      const float alpha = fast_exp2(mrow[r] - mn);
      const float p0 = mk0 ? fast_exp2(a0 - mn) : 0.f;
      const float p1 = mk1 ? fast_exp2(a1 - mn) : 0.f;
      float rs = p0 + p1;
      #pragma unroll
      for (int msk = 1; msk < 16; msk <<= 1) rs += __shfl_xor(rs, msk, 32);
      lrow[r] = lrow[r] * alpha + rs;
      mrow[r] = mn;
      acc0[r] *= alpha; acc1[r] *= alpha; acc2[r] *= alpha; acc3[r] *= alpha;
      const int row = r + 8 * hi;
      psm[row * Bc + ln]      = (half_t)p0;
      psm[row * Bc + 16 + ln] = (half_t)p1;
    }
    asm volatile("s_wait_dscnt 0" ::: "memory");
    v16h aP;
    {
      const half_t* pp = psm + ln * Bc + hi * 16;
      #pragma unroll
      for (int e = 0; e < 8; ++e) { aP[e] = pp[e]; aP[8 + e] = pp[8 + e]; }
    }
    v16h bv0, bv1, bv2, bv3;
    {
      const half_t* vq0 = vsm + (0 * 16 + ln) * Bc + hi * 16;
      const half_t* vq1 = vsm + (1 * 16 + ln) * Bc + hi * 16;
      const half_t* vq2 = vsm + (2 * 16 + ln) * Bc + hi * 16;
      const half_t* vq3 = vsm + (3 * 16 + ln) * Bc + hi * 16;
      #pragma unroll
      for (int e = 0; e < 16; ++e) { bv0[e] = vq0[e]; bv1[e] = vq1[e]; bv2[e] = vq2[e]; bv3[e] = vq3[e]; }
    }
    acc0 = __builtin_amdgcn_wmma_f32_16x16x32_f16(false, aP, false, bv0, (short)0, acc0, false, false);
    acc1 = __builtin_amdgcn_wmma_f32_16x16x32_f16(false, aP, false, bv1, (short)0, acc1, false, false);
    acc2 = __builtin_amdgcn_wmma_f32_16x16x32_f16(false, aP, false, bv2, (short)0, acc2, false, false);
    acc3 = __builtin_amdgcn_wmma_f32_16x16x32_f16(false, aP, false, bv3, (short)0, acc3, false, false);
    asm volatile("" ::: "memory");
  }
  #pragma unroll
  for (int r = 0; r < 8; ++r) {
    const int row = r + 8 * hi;
    const float inv = lrow[r] > 0.f ? 1.f / lrow[r] : 0.f;
    float* op = O + (((size_t)b * Sq + (q0 + row)) * Hh + h) * Dd;
    op[0 * 16 + ln] = acc0[r] * inv;
    op[1 * 16 + ln] = acc1[r] * inv;
    op[2 * 16 + ln] = acc2[r] * inv;
    op[3 * 16 + ln] = acc3[r] * inv;
  }
}

extern "C" void kernel_launch(void* const* d_in, const int* in_sizes, int n_in,
                              void* d_out, int out_size, void* d_ws, size_t ws_size,
                              hipStream_t stream) {
  (void)in_sizes; (void)n_in; (void)out_size;
  const float* Q  = (const float*)d_in[0];
  const float* K  = (const float*)d_in[1];
  const float* V  = (const float*)d_in[2];
  const int*   Mk = (const int*)d_in[3];
  float* out = (float*)d_out;

  const size_t wsNeeded = 2 * KV_ELEMS * sizeof(half_t);   // 16 MiB
  if (d_ws != nullptr && ws_size >= wsNeeded) {
    half_t* Kh = (half_t*)d_ws;
    half_t* Vh = Kh + KV_ELEMS;
    prep_k_kernel<<<dim3((unsigned)(KV_ELEMS / 256)), 256, 0, stream>>>(K, Kh);
    prep_v_kernel<<<dim3(Sq / 64, Hh, Bz), 256, 0, stream>>>(V, Vh);
    fa_fwd_f16kv_kernel<<<dim3(Sq / 64, Hh, Bz), 128, 0, stream>>>(Q, Kh, Vh, Mk, out);
  } else {
    fa_fwd_legacy_kernel<<<dim3(Sq / 64, Hh, Bz), 128, 0, stream>>>(Q, K, V, Mk, out);
  }
}
